// GraphTransformer_22359599743565
// MI455X (gfx1250) — matrix operand order repaired and checked by reading, not compiled, not run on hardware
//
#include <hip/hip_runtime.h>

// Graph transformer forward for MI455X (gfx1250, wave32).
// All GEMM-shaped work via V_WMMA_F32_16X16X32_F16 (f16 operands staged in
// workspace with fragment-friendly layouts; f32 accumulate).
// Needs ~25 MB of workspace.

#define NFEAT 128
#define EFEAT 64
#define HID   256
#define NHEAD 8
#define HDIM  32
#define FFD   1024
#define NLAYER 4
#define OUTD  1280

typedef __attribute__((ext_vector_type(16))) _Float16 v16h;
typedef __attribute__((ext_vector_type(8)))  _Float16 v8h;
typedef __attribute__((ext_vector_type(8)))  float    v8f;

__device__ __forceinline__ v8f wmma_f16(v16h a, v16h b, v8f c) {
  // 8 args: (neg_a, A, neg_b, B, c_mod, C, reuse_a, reuse_b)
  return __builtin_amdgcn_wmma_f32_16x16x32_f16(false, a, false, b, (short)0, c,
                                                false, false);
}

__device__ __forceinline__ v16h cat8(v8h lo, v8h hi) {
  return __builtin_shufflevector(lo, hi, 0, 1, 2, 3, 4, 5, 6, 7,
                                 8, 9, 10, 11, 12, 13, 14, 15);
}

// A-fragment (16x32 f16): lane half h, elements 0..15 <- K=16h..16h+15
// -> two contiguous v8h loads.
__device__ __forceinline__ v16h load_afrag(const _Float16* rowp, int k0, int half) {
  v8h lo = *(const v8h*)(rowp + k0 + half * 16);
  v8h hi = *(const v8h*)(rowp + k0 + half * 16 + 8);
  return cat8(lo, hi);
}

// A-fragment built from f32 data (LDS attention scores).
__device__ __forceinline__ v16h load_afrag_f32(const float* rowp, int k0, int half) {
  v16h a;
#pragma unroll
  for (int i = 0; i < 8; ++i) {
    a[i]     = (_Float16)rowp[k0 + half * 16 + i];
    a[8 + i] = (_Float16)rowp[k0 + half * 16 + 8 + i];
  }
  return a;
}

// ---------------------------------------------------------------------------
// f32 -> f16 conversions (operand staging)
// ---------------------------------------------------------------------------
__global__ void k_cvt(const float* __restrict__ in, _Float16* __restrict__ out, int n) {
  int i = blockIdx.x * blockDim.x + threadIdx.x;
  if (i < n) out[i] = (_Float16)in[i];
}

// B[K,Nc] f32 row-major -> Bt[Nc,K] f16 (so B-fragments are contiguous in K)
__global__ void k_cvtT(const float* __restrict__ in, _Float16* __restrict__ out,
                       int K, int Nc) {
  int i = blockIdx.x * blockDim.x + threadIdx.x;
  if (i >= K * Nc) return;
  int k = i / Nc, ncol = i % Nc;
  out[(size_t)ncol * K + k] = (_Float16)in[i];
}

// ---------------------------------------------------------------------------
// WMMA GEMM: C = act(A[M,K] @ B + bias) (+resid). A: f16 row-major,
// Bt: f16 [Nc,K]. One wave computes a 16x64 strip (1 A-frag -> 4 WMMAs).
// Optional f32 out C, optional f16 out C16 (row-major or transposed [Nc,M]).
// act: 0=none 1=gelu 2=tanh. M%16==0, Nc%64==0, K%32==0.
// ---------------------------------------------------------------------------
__global__ __launch_bounds__(256) void k_gemm(
    const _Float16* __restrict__ A, const _Float16* __restrict__ Bt,
    const float* __restrict__ bias, const float* __restrict__ resid,
    float* __restrict__ C, _Float16* __restrict__ C16,
    int M, int K, int Nc, int act, int transpose16) {
  int waveId = threadIdx.x >> 5;
  int lane = threadIdx.x & 31;
  int half = lane >> 4, l16 = lane & 15;
  int groupsN = Nc >> 6;
  int grp = blockIdx.x * 8 + waveId;
  if (grp >= (M >> 4) * groupsN) return;
  int tm = grp / groupsN, tg = grp % groupsN;

  const _Float16* Ar = A + (size_t)(tm * 16 + l16) * K;
  const _Float16* B0 = Bt + (size_t)(tg * 64 + l16) * K;
  v8f acc[4] = {{}, {}, {}, {}};
  for (int k0 = 0; k0 < K; k0 += 32) {
    __builtin_prefetch(Ar + k0 + 128, 0, 3);           // global_prefetch_b8
    v16h a = load_afrag(Ar, k0, half);
#pragma unroll
    for (int t = 0; t < 4; ++t) {
      v16h b = *(const v16h*)(B0 + (size_t)(t * 16) * K + k0 + half * 16);
      acc[t] = wmma_f16(a, b, acc[t]);
    }
  }
#pragma unroll
  for (int t = 0; t < 4; ++t) {
    int ncol = tg * 64 + t * 16 + l16;
    float bv = bias ? bias[ncol] : 0.f;
#pragma unroll
    for (int r = 0; r < 8; ++r) {                      // C: VGPR r -> row r (+8 hi)
      int m = tm * 16 + half * 8 + r;
      float v = acc[t][r] + bv;
      if (act == 1)      v = 0.5f * v * (1.f + erff(v * 0.70710678f));
      else if (act == 2) v = tanhf(v);
      if (resid) v += resid[(size_t)m * Nc + ncol];
      if (C)   C[(size_t)m * Nc + ncol] = v;
      if (C16) {
        if (transpose16) C16[(size_t)ncol * M + m] = (_Float16)v;
        else             C16[(size_t)m * Nc + ncol] = (_Float16)v;
      }
    }
  }
}

// ---------------------------------------------------------------------------
// Adjacency bitmask + CSR build
// ---------------------------------------------------------------------------
__global__ void k_zero_u32(unsigned* p, int n) {
  int i = blockIdx.x * blockDim.x + threadIdx.x;
  if (i < n) p[i] = 0u;
}

__global__ void k_adj_set(const int* __restrict__ src, const int* __restrict__ dst,
                          unsigned* __restrict__ adj, int E, int n) {
  int id = blockIdx.x * blockDim.x + threadIdx.x;
  int words = n >> 5;
  if (id < E) {
    int s = src[id], d = dst[id];
    atomicOr(&adj[(size_t)s * words + (d >> 5)], 1u << (d & 31));
    atomicOr(&adj[(size_t)d * words + (s >> 5)], 1u << (s & 31));
  } else if (id < E + n) {
    int i = id - E;  // self loops
    atomicOr(&adj[(size_t)i * words + (i >> 5)], 1u << (i & 31));
  }
}

__global__ void k_hist(const int* __restrict__ src, int* __restrict__ cnt, int E) {
  int e = blockIdx.x * blockDim.x + threadIdx.x;
  if (e < E) atomicAdd(&cnt[src[e]], 1);
}

__global__ void k_scan(const int* __restrict__ cnt, int* __restrict__ rowptr, int n) {
  if (blockIdx.x == 0 && threadIdx.x == 0) {
    int acc = 0;
    for (int i = 0; i < n; ++i) { rowptr[i] = acc; acc += cnt[i]; }
    rowptr[n] = acc;
  }
}

__global__ void k_fill(const int* __restrict__ src, int* __restrict__ cnt2,
                       const int* __restrict__ rowptr, int* __restrict__ csre, int E) {
  int e = blockIdx.x * blockDim.x + threadIdx.x;
  if (e >= E) return;
  int s = src[e];
  int pos = atomicAdd(&cnt2[s], 1);
  csre[rowptr[s] + pos] = e;
}

// ---------------------------------------------------------------------------
// Folded edge bias: Wcomb[64,8] = W_edge @ Wep[i]; bcomb = b_edge@Wep[i]+bep[i]
// (all linear => exact); then ebias[E,8] = edge_features @ Wcomb + bcomb.
// ---------------------------------------------------------------------------
__global__ void k_ebias_w(const float* __restrict__ W_edge, const float* __restrict__ b_edge,
                          const float* __restrict__ Wep, const float* __restrict__ bep,
                          float* __restrict__ Wcomb, float* __restrict__ bcomb) {
  int tid = threadIdx.x;
  for (int idx = tid; idx < EFEAT * NHEAD; idx += 256) {
    int k = idx / NHEAD, h = idx % NHEAD;
    float s = 0.f;
    for (int j = 0; j < HID; ++j) s += W_edge[k * HID + j] * Wep[j * NHEAD + h];
    Wcomb[idx] = s;
  }
  if (tid < NHEAD) {
    float s = bep[tid];
    for (int j = 0; j < HID; ++j) s += b_edge[j] * Wep[j * NHEAD + tid];
    bcomb[tid] = s;
  }
}

__global__ void k_ebias(const float* __restrict__ ef, const float* __restrict__ Wcomb,
                        const float* __restrict__ bcomb, float* __restrict__ ebias, int E) {
  int id = blockIdx.x * blockDim.x + threadIdx.x;
  if (id >= E * NHEAD) return;
  int e = id / NHEAD, h = id % NHEAD;
  const float* row = ef + (size_t)e * EFEAT;
  float s = bcomb[h];
  for (int k = 0; k < EFEAT; ++k) s += row[k] * Wcomb[k * NHEAD + h];
  ebias[id] = s;
}

// ---------------------------------------------------------------------------
// Attention: block = (16 query rows, 1 head). LDS score strip 16 x n (f32).
// Q16,K16: f16 [n,HID] row-major; Vt16: f16 [HID,n] (pre-transposed by V GEMM)
// so every WMMA B-fragment is one contiguous 32-byte load.
// ---------------------------------------------------------------------------
__global__ __launch_bounds__(256) void k_attn(
    const _Float16* __restrict__ Q16, const _Float16* __restrict__ K16,
    const _Float16* __restrict__ Vt16, const float* __restrict__ ebias,
    const int* __restrict__ rowptr, const int* __restrict__ csre,
    const int* __restrict__ dsti, const unsigned* __restrict__ adj,
    _Float16* __restrict__ ctx16, int n) {
  extern __shared__ float smem[];
  float* sc   = smem;               // [16][n] scores
  float* part = smem + 16 * n;      // [8][16][32] cross-wave partials
  int tid = threadIdx.x;
  int w = tid >> 5, lane = tid & 31, half = lane >> 4, l16 = lane & 15;
  int qb = blockIdx.x, h = blockIdx.y;
  const float invs = 0.17677669529663687f;   // 1/sqrt(HDIM)

  // --- Phase 1: scores = Q K^T / sqrt(d), one WMMA per 16-key tile ---
  v16h aq = load_afrag(Q16 + (size_t)(qb * 16 + l16) * HID + h * HDIM, 0, half);
  int ntiles = n >> 4;
  for (int jt = w; jt < ntiles; jt += 8) {
    const _Float16* kp = K16 + (size_t)(jt * 16 + l16) * HID + h * HDIM + half * 16;
    __builtin_prefetch(kp + 8 * 16 * HID, 0, 3);
    v16h b = *(const v16h*)kp;
    v8f s = {};
    s = wmma_f16(aq, b, s);
#pragma unroll
    for (int r = 0; r < 8; ++r)
      sc[(half * 8 + r) * n + jt * 16 + l16] = s[r] * invs;
  }
  __syncthreads();

  // --- Phase 2: scatter edge bias (scores[src,dst] += ebias[e,h]) via CSR ---
  {
    int r = tid >> 4, t = tid & 15;
    int grow = qb * 16 + r;
    int pe = rowptr[grow + 1];
    for (int p = rowptr[grow] + t; p < pe; p += 16) {
      int e = csre[p];
      atomicAdd(&sc[r * n + dsti[e]], ebias[e * NHEAD + h]);
    }
  }
  __syncthreads();

  // --- Phase 3: adjacency-masked softmax per row (16 lanes/row, wave32) ---
  {
    int r = tid >> 4, t = tid & 15;
    int grow = qb * 16 + r;
    const unsigned* arow = adj + (size_t)grow * (n >> 5);
    float m = -1e30f;
    for (int c = t; c < n; c += 16)
      if ((arow[c >> 5] >> (c & 31)) & 1u) m = fmaxf(m, sc[r * n + c]);
#pragma unroll
    for (int d = 8; d >= 1; d >>= 1) m = fmaxf(m, __shfl_xor(m, d));
    float sum = 0.f;
    for (int c = t; c < n; c += 16) {
      float p = 0.f;
      if ((arow[c >> 5] >> (c & 31)) & 1u) p = __expf(sc[r * n + c] - m);
      sc[r * n + c] = p;
      sum += p;
    }
#pragma unroll
    for (int d = 8; d >= 1; d >>= 1) sum += __shfl_xor(sum, d);
    float inv = 1.f / sum;   // self-loop guarantees sum > 0
    for (int c = t; c < n; c += 16) sc[r * n + c] *= inv;
  }
  __syncthreads();

  // --- Phase 4: ctx = attn @ V (head slice), 2 WMMAs per 32-key chunk ---
  {
    v8f acc0 = {}, acc1 = {};
    int nchunks = n >> 5;
    for (int kc = w; kc < nchunks; kc += 8) {
      v16h a = load_afrag_f32(sc + l16 * n, kc * 32, half);
      const _Float16* vp0 = Vt16 + (size_t)(h * HDIM + l16) * n + kc * 32 + half * 16;
      const _Float16* vp1 = vp0 + (size_t)16 * n;
      v16h b0 = *(const v16h*)vp0;
      v16h b1 = *(const v16h*)vp1;
      acc0 = wmma_f16(a, b0, acc0);
      acc1 = wmma_f16(a, b1, acc1);
    }
#pragma unroll
    for (int r = 0; r < 8; ++r) {
      part[(w * 16 + half * 8 + r) * 32 + l16]      = acc0[r];
      part[(w * 16 + half * 8 + r) * 32 + 16 + l16] = acc1[r];
    }
  }
  __syncthreads();
  for (int idx = tid; idx < 16 * 32; idx += 256) {
    int r = idx >> 5, d = idx & 31;
    float s = 0.f;
#pragma unroll
    for (int ww = 0; ww < 8; ++ww) s += part[(ww * 16 + r) * 32 + d];
    ctx16[(size_t)(qb * 16 + r) * HID + h * HDIM + d] = (_Float16)s;
  }
}

// ---------------------------------------------------------------------------
// LayerNorm (optional residual add), in place on f32 x; also emits f16 copy.
// ---------------------------------------------------------------------------
__global__ __launch_bounds__(HID) void k_ln(float* __restrict__ x,
                                            const float* __restrict__ add,
                                            const float* __restrict__ g,
                                            const float* __restrict__ b,
                                            _Float16* __restrict__ x16) {
  int row = blockIdx.x, tid = threadIdx.x;
  __shared__ float red[HID];
  float v = x[(size_t)row * HID + tid];
  if (add) v += add[(size_t)row * HID + tid];
  red[tid] = v;
  __syncthreads();
  for (int s = HID / 2; s > 0; s >>= 1) {
    if (tid < s) red[tid] += red[tid + s];
    __syncthreads();
  }
  float mean = red[0] * (1.f / HID);
  __syncthreads();
  float d = v - mean;
  red[tid] = d * d;
  __syncthreads();
  for (int s = HID / 2; s > 0; s >>= 1) {
    if (tid < s) red[tid] += red[tid + s];
    __syncthreads();
  }
  float var = red[0] * (1.f / HID);
  float o = d * rsqrtf(var + 1e-5f) * g[tid] + b[tid];
  x[(size_t)row * HID + tid] = o;
  x16[(size_t)row * HID + tid] = (_Float16)o;
}

// ---------------------------------------------------------------------------
// Global pooling: mean / max / softmax-attention, one block.
// ---------------------------------------------------------------------------
__global__ __launch_bounds__(256) void k_pool(const float* __restrict__ x,
                                              const float* __restrict__ t,
                                              const float* __restrict__ Wp2,
                                              const float* __restrict__ bp2,
                                              float* __restrict__ g, int n) {
  __shared__ float pl[2048];
  __shared__ float red[256];
  int tid = threadIdx.x;
  for (int row = tid; row < n; row += 256) {
    float s = 0.f;
    for (int k = 0; k < HID; ++k) s += t[(size_t)row * HID + k] * Wp2[k];
    pl[row] = s + bp2[0];
  }
  __syncthreads();
  float lm = -1e30f;
  for (int row = tid; row < n; row += 256) lm = fmaxf(lm, pl[row]);
  red[tid] = lm; __syncthreads();
  for (int s = 128; s > 0; s >>= 1) { if (tid < s) red[tid] = fmaxf(red[tid], red[tid + s]); __syncthreads(); }
  lm = red[0]; __syncthreads();
  float ls = 0.f;
  for (int row = tid; row < n; row += 256) { float p = __expf(pl[row] - lm); pl[row] = p; ls += p; }
  red[tid] = ls; __syncthreads();
  for (int s = 128; s > 0; s >>= 1) { if (tid < s) red[tid] += red[tid + s]; __syncthreads(); }
  ls = red[0]; __syncthreads();
  int col = tid;  // 256 threads == HID columns
  float sm = 0.f, mx = -1e30f, ap = 0.f;
  for (int row = 0; row < n; ++row) {
    float v = x[(size_t)row * HID + col];
    sm += v; mx = fmaxf(mx, v); ap += v * pl[row];
  }
  g[col] = sm / n;
  g[HID + col] = mx;
  g[2 * HID + col] = ap / ls;
}

__global__ __launch_bounds__(512) void k_head1(const float* __restrict__ g,
                                               const float* __restrict__ Wo1,
                                               const float* __restrict__ bo1,
                                               float* __restrict__ o1) {
  int j = threadIdx.x;  // 512
  float s = bo1[j];
  for (int k = 0; k < 3 * HID; ++k) s += g[k] * Wo1[k * 2 * HID + j];
  o1[j] = fmaxf(s, 0.f);
}

__global__ void k_head2(const float* __restrict__ o1, const float* __restrict__ Wo2,
                        const float* __restrict__ bo2, float* __restrict__ out) {
  int j = blockIdx.x * blockDim.x + threadIdx.x;
  if (j >= OUTD) return;
  float s = bo2[j];
  for (int k = 0; k < 2 * HID; ++k) s += o1[k] * Wo2[k * OUTD + j];
  out[j] = s;
}

// ---------------------------------------------------------------------------
extern "C" void kernel_launch(void* const* d_in, const int* in_sizes, int n_in,
                              void* d_out, int out_size, void* d_ws, size_t ws_size,
                              hipStream_t stream) {
  const float* node_features = (const float*)d_in[0];
  const float* edge_features = (const float*)d_in[1];
  const int*   edge_index    = (const int*)d_in[2];
  const float* W_node = (const float*)d_in[3];
  const float* b_node = (const float*)d_in[4];
  const float* W_edge = (const float*)d_in[5];
  const float* b_edge = (const float*)d_in[6];
  const float* pos_emb = (const float*)d_in[7];
  const float* Wq = (const float*)d_in[8];   const float* bq = (const float*)d_in[9];
  const float* Wk = (const float*)d_in[10];  const float* bk = (const float*)d_in[11];
  const float* Wv = (const float*)d_in[12];  const float* bv = (const float*)d_in[13];
  const float* Wo = (const float*)d_in[14];  const float* bo = (const float*)d_in[15];
  const float* Wep = (const float*)d_in[16]; const float* bep = (const float*)d_in[17];
  const float* Wf1 = (const float*)d_in[18]; const float* bf1 = (const float*)d_in[19];
  const float* Wf2 = (const float*)d_in[20]; const float* bf2 = (const float*)d_in[21];
  const float* g1 = (const float*)d_in[22];  const float* be1 = (const float*)d_in[23];
  const float* g2 = (const float*)d_in[24];  const float* be2 = (const float*)d_in[25];
  const float* g_ln = (const float*)d_in[26]; const float* b_ln = (const float*)d_in[27];
  const float* Wp1 = (const float*)d_in[28]; const float* bp1 = (const float*)d_in[29];
  const float* Wp2 = (const float*)d_in[30]; const float* bp2 = (const float*)d_in[31];
  const float* Wo1 = (const float*)d_in[32]; const float* bo1 = (const float*)d_in[33];
  const float* Wo2 = (const float*)d_in[34]; const float* bo2 = (const float*)d_in[35];
  (void)n_in; (void)out_size; (void)ws_size;

  int n = in_sizes[0] / NFEAT;   // 2048
  int E = in_sizes[1] / EFEAT;   // 65536
  const int* src = edge_index;
  const int* dst = edge_index + E;

  // workspace carve-up (256-byte aligned slices)
  char* ws = (char*)d_ws;
  auto alloc = [&](size_t bytes) {
    char* p = ws;
    ws += (bytes + 255) & ~(size_t)255;
    return p;
  };
  // f32 state
  float* x    = (float*)alloc((size_t)n * HID * 4);
  float* ab   = (float*)alloc((size_t)n * HID * 4);
  float* tb   = (float*)alloc((size_t)n * HID * 4);
  float* eb   = (float*)alloc((size_t)E * NHEAD * 4);
  // f16 activations (fragment-friendly layouts)
  _Float16* nf16  = (_Float16*)alloc((size_t)n * NFEAT * 2);
  _Float16* x16   = (_Float16*)alloc((size_t)n * HID * 2);
  _Float16* Q16   = (_Float16*)alloc((size_t)n * HID * 2);
  _Float16* K16   = (_Float16*)alloc((size_t)n * HID * 2);
  _Float16* Vt16  = (_Float16*)alloc((size_t)n * HID * 2);   // [HID, n]
  _Float16* ctx16 = (_Float16*)alloc((size_t)n * HID * 2);
  _Float16* ff16  = (_Float16*)alloc((size_t)n * FFD * 2);
  // f16 transposed weights [Nc, K]
  _Float16* WnodeT = (_Float16*)alloc((size_t)NFEAT * HID * 2);
  _Float16* WqT = (_Float16*)alloc((size_t)NLAYER * HID * HID * 2);
  _Float16* WkT = (_Float16*)alloc((size_t)NLAYER * HID * HID * 2);
  _Float16* WvT = (_Float16*)alloc((size_t)NLAYER * HID * HID * 2);
  _Float16* WoT = (_Float16*)alloc((size_t)NLAYER * HID * HID * 2);
  _Float16* Wf1T = (_Float16*)alloc((size_t)NLAYER * HID * FFD * 2);
  _Float16* Wf2T = (_Float16*)alloc((size_t)NLAYER * FFD * HID * 2);
  _Float16* Wp1T = (_Float16*)alloc((size_t)HID * HID * 2);
  // graph structure
  unsigned* adj = (unsigned*)alloc((size_t)n * (n / 32) * 4);
  int* rowptr = (int*)alloc((size_t)(n + 1) * 4);
  int* cnt    = (int*)alloc((size_t)n * 4);
  int* cnt2   = (int*)alloc((size_t)n * 4);
  int* csre   = (int*)alloc((size_t)E * 4);
  float* Wcomb = (float*)alloc(EFEAT * NHEAD * 4);
  float* bcomb = (float*)alloc(NHEAD * 4);
  float* gvec  = (float*)alloc(3 * HID * 4);
  float* o1    = (float*)alloc(2 * HID * 4);

  // ---- operand staging: inputs + all GEMM weights to f16 ----
  k_cvt<<<(n * NFEAT + 255) / 256, 256, 0, stream>>>(node_features, nf16, n * NFEAT);
  k_cvtT<<<(NFEAT * HID + 255) / 256, 256, 0, stream>>>(W_node, WnodeT, NFEAT, HID);
  for (int i = 0; i < NLAYER; ++i) {
    size_t oH = (size_t)i * HID * HID;
    k_cvtT<<<(HID * HID + 255) / 256, 256, 0, stream>>>(Wq + oH, WqT + oH, HID, HID);
    k_cvtT<<<(HID * HID + 255) / 256, 256, 0, stream>>>(Wk + oH, WkT + oH, HID, HID);
    k_cvtT<<<(HID * HID + 255) / 256, 256, 0, stream>>>(Wv + oH, WvT + oH, HID, HID);
    k_cvtT<<<(HID * HID + 255) / 256, 256, 0, stream>>>(Wo + oH, WoT + oH, HID, HID);
    size_t oF = (size_t)i * HID * FFD;
    k_cvtT<<<(HID * FFD + 255) / 256, 256, 0, stream>>>(Wf1 + oF, Wf1T + oF, HID, FFD);
    k_cvtT<<<(HID * FFD + 255) / 256, 256, 0, stream>>>(Wf2 + oF, Wf2T + oF, FFD, HID);
  }
  k_cvtT<<<(HID * HID + 255) / 256, 256, 0, stream>>>(Wp1, Wp1T, HID, HID);

  // node encode: x = node_features @ W_node + b_node + pos_emb[:n]  (f32 + f16)
  int grpH = (n / 16) * (HID / 64);
  int grpF = (n / 16) * (FFD / 64);
  k_gemm<<<(grpH + 7) / 8, 256, 0, stream>>>(nf16, WnodeT, b_node, pos_emb,
                                             x, x16, n, NFEAT, HID, 0, 0);

  // adjacency bitmask + CSR by source (structure is layer-invariant)
  int adjWords = n * (n / 32);
  k_zero_u32<<<(adjWords + 255) / 256, 256, 0, stream>>>(adj, adjWords);
  k_zero_u32<<<(n + 255) / 256, 256, 0, stream>>>((unsigned*)cnt, n);
  k_zero_u32<<<(n + 255) / 256, 256, 0, stream>>>((unsigned*)cnt2, n);
  k_adj_set<<<(E + n + 255) / 256, 256, 0, stream>>>(src, dst, adj, E, n);
  k_hist<<<(E + 255) / 256, 256, 0, stream>>>(src, cnt, E);
  k_scan<<<1, 1, 0, stream>>>(cnt, rowptr, n);
  k_fill<<<(E + 255) / 256, 256, 0, stream>>>(src, cnt2, rowptr, csre, E);

  size_t attn_lds = (size_t)(16 * n + 8 * 16 * 32) * sizeof(float);  // 144 KB < 320 KB/WGP
  hipFuncSetAttribute(reinterpret_cast<const void*>(k_attn),
                      hipFuncAttributeMaxDynamicSharedMemorySize, (int)attn_lds);

  for (int i = 0; i < NLAYER; ++i) {
    size_t oH = (size_t)i * HID * HID;
    size_t oF = (size_t)i * HID * FFD;
    // QKV: f16-only outputs; V is written pre-transposed [HID, n]
    k_gemm<<<(grpH + 7) / 8, 256, 0, stream>>>(x16, WqT + oH, bq + i * HID, nullptr,
                                               nullptr, Q16, n, HID, HID, 0, 0);
    k_gemm<<<(grpH + 7) / 8, 256, 0, stream>>>(x16, WkT + oH, bk + i * HID, nullptr,
                                               nullptr, K16, n, HID, HID, 0, 0);
    k_gemm<<<(grpH + 7) / 8, 256, 0, stream>>>(x16, WvT + oH, bv + i * HID, nullptr,
                                               nullptr, Vt16, n, HID, HID, 0, 1);

    // folded edge bias for this layer
    k_ebias_w<<<1, 256, 0, stream>>>(W_edge, b_edge, Wep + (size_t)i * HID * NHEAD,
                                     bep + i * NHEAD, Wcomb, bcomb);
    k_ebias<<<(E * NHEAD + 255) / 256, 256, 0, stream>>>(edge_features, Wcomb, bcomb, eb, E);

    k_attn<<<dim3(n / 16, NHEAD), 256, attn_lds, stream>>>(Q16, K16, Vt16, eb, rowptr,
                                                           csre, dst, adj, ctx16, n);

    k_gemm<<<(grpH + 7) / 8, 256, 0, stream>>>(ctx16, WoT + oH, bo + i * HID, nullptr,
                                               ab, nullptr, n, HID, HID, 0, 0);
    k_ln<<<n, HID, 0, stream>>>(x, ab, g1 + i * HID, be1 + i * HID, x16);

    k_gemm<<<(grpF + 7) / 8, 256, 0, stream>>>(x16, Wf1T + oF, bf1 + i * FFD, nullptr,
                                               nullptr, ff16, n, HID, FFD, 1, 0);
    k_gemm<<<(grpH + 7) / 8, 256, 0, stream>>>(ff16, Wf2T + oF, bf2 + i * HID, nullptr,
                                               ab, nullptr, n, FFD, HID, 0, 0);
    k_ln<<<n, HID, 0, stream>>>(x, ab, g2 + i * HID, be2 + i * HID, x16);
  }

  k_ln<<<n, HID, 0, stream>>>(x, nullptr, g_ln, b_ln, x16);
  k_gemm<<<(grpH + 7) / 8, 256, 0, stream>>>(x16, Wp1T, bp1, nullptr,
                                             tb, nullptr, n, HID, HID, 2, 0);
  k_pool<<<1, 256, 0, stream>>>(x, tb, Wp2, bp2, gvec, n);
  k_head1<<<1, 512, 0, stream>>>(gvec, Wo1, bo1, o1);
  k_head2<<<(OUTD + 255) / 256, 256, 0, stream>>>(o1, Wo2, bo2, (float*)d_out);
}
